// PFCAttention_11166914969914
// MI455X (gfx1250) — matrix operand order repaired and checked by reading, compiled, not run on hardware
//
#include <hip/hip_runtime.h>
#include <hip/hip_bf16.h>

// ---------------------------------------------------------------------------
// CDNA5 (gfx1250) PFC attention, v2:
//  - all intermediates bf16 in global memory (one conversion per element)
//  - GEMM LDS staging via Tensor Data Mover (tensor_load_to_lds, TENSORcnt),
//    double-buffered; weights pre-transposed so both operands are [rows x K]
//  - all matrix math on v_wmma_f32_16x16x32_bf16 (f32 accumulate)
// ---------------------------------------------------------------------------

typedef __attribute__((ext_vector_type(16))) __bf16        v16bf;
typedef __attribute__((ext_vector_type(8)))  float         v8f;
typedef __attribute__((ext_vector_type(4)))  unsigned int  v4u;
typedef __attribute__((ext_vector_type(8)))  unsigned int  v8u;

union FragBF {
    v16bf          v;
    unsigned short us[16];
    uint4          q[2];
};

__device__ __forceinline__ unsigned short f2bf(float f) {
    unsigned int u = __float_as_uint(f);
    unsigned int r = u + 0x7FFFu + ((u >> 16) & 1u);   // round-to-nearest-even
    return (unsigned short)(r >> 16);
}
__device__ __forceinline__ float bf2f(unsigned short u) {
    return __uint_as_float(((unsigned)u) << 16);
}

__device__ __forceinline__ v8f wmma_bf16(const v16bf& a, const v16bf& b, const v8f& c) {
    return __builtin_amdgcn_wmma_f32_16x16x32_bf16(
        false, a, false, b, (short)0, c, false, false);
}

// ---------------------------------------------------------------------------
// Tensor Data Mover: 2D bf16 tile (tile_d1 rows x tile_d0 elems) global->LDS,
// with LDS row padding 64B data + 16B pad (pitch 80B == LSTRIDE*2).
// Descriptor per CDNA5 ISA. Wave-uniform args.
// ---------------------------------------------------------------------------
__device__ __forceinline__ void tdm_load_2d(unsigned lds_byte_addr,
                                            const unsigned short* gptr,
                                            unsigned tensor_d0,   // row length (elems)
                                            unsigned tensor_d1,   // total rows
                                            unsigned stride_e,    // row stride (elems)
                                            unsigned tile_d0,     // tile cols (elems)
                                            unsigned tile_d1)     // tile rows
{
    unsigned long long ga = (unsigned long long)(const void*)gptr;
    v4u g0;
    g0[0] = 1u;                                           // count=1, user mode
    g0[1] = lds_byte_addr;                                // lds_addr
    g0[2] = (unsigned)(ga & 0xFFFFFFFFu);                 // global_addr[31:0]
    g0[3] = (unsigned)((ga >> 32) & 0x1FFFFFFu)           // global_addr[56:32]
          | (2u << 30);                                   // type = 2 ("image")
    v8u g1;
    g1[0] = (1u << 16)      // data_size = 1 -> 2 bytes
          | (1u << 20)      // pad_enable
          | (3u << 22)      // pad_interval = 16 DWORDs (64B, one tile row)
          | (3u << 25);     // pad_amount   = 4 DWORDs (16B)
    g1[1] = (tensor_d0 & 0xFFFFu) << 16;                          // dim0[15:0]
    g1[2] = ((tensor_d0 >> 16) & 0xFFFFu) | ((tensor_d1 & 0xFFFFu) << 16);
    g1[3] = ((tensor_d1 >> 16) & 0xFFFFu) | ((tile_d0 & 0xFFFFu) << 16);
    g1[4] = tile_d1 & 0xFFFFu;                                    // tile_dim2 = 0
    g1[5] = stride_e;                                             // dim0_stride[31:0]
    g1[6] = 0;                                                    // dim0_stride[47:32]
    g1[7] = 0;                                                    // dim1_stride unused (2D)
    asm volatile("tensor_load_to_lds %0, %1" :: "s"(g0), "s"(g1) : "memory");
}

// ===========================================================================
// Elementwise f32 -> bf16 (4 elems / thread)
// ===========================================================================
__global__ __launch_bounds__(256)
void cvt_bf16(const float* __restrict__ src, unsigned short* __restrict__ dst,
              long long n4)
{
    long long t = (long long)blockIdx.x * blockDim.x + threadIdx.x;
    if (t >= n4) return;
    float4 v = *(const float4*)(src + t * 4);
    uint2 pk;
    pk.x = (unsigned)f2bf(v.x) | ((unsigned)f2bf(v.y) << 16);
    pk.y = (unsigned)f2bf(v.z) | ((unsigned)f2bf(v.w) << 16);
    *(uint2*)(dst + t * 4) = pk;
}

// Transpose + convert: src f32 [K x N] -> dst bf16 [N x K]
__global__ __launch_bounds__(256)
void transpose_cvt(const float* __restrict__ src, unsigned short* __restrict__ dst,
                   int K, int N)
{
    int t = blockIdx.x * blockDim.x + threadIdx.x;
    if (t >= K * N) return;
    int n = t / K;
    int k = t - n * K;
    dst[t] = f2bf(src[(size_t)k * N + n]);
}

// ===========================================================================
// GEMM: C[M,N] = A[M,K] @ Bt[N,K]^T + bias[N]; A,Bt bf16, acc f32.
// Block 128x128x32, 256 thr (8 waves 2x4), wave tile 64x32.
// Tiles staged by TDM (wave 0), double-buffered on TENSORcnt.
// ===========================================================================
#define BM 128
#define BN 128
#define BK 32
#define LSTRIDE 40   // 32 bf16 + 8 pad: 80B pitch (matches TDM pad descriptor)

__global__ __launch_bounds__(256)
void gemm_bf16(const unsigned short* __restrict__ A,
               const unsigned short* __restrict__ Bt,
               const float* __restrict__ bias, void* __restrict__ Cout,
               int M, int N, int K, int lda, int ldb, int ldc, int f32out)
{
    __shared__ unsigned short sA[2][BM * LSTRIDE];
    __shared__ unsigned short sB[2][BN * LSTRIDE];

    const int tid   = threadIdx.x;
    const int lane  = tid & 31;
    const int wid   = tid >> 5;
    const int wm    = (wid >> 2) * 64;
    const int wn    = (wid & 3) * 32;
    const int l16   = lane & 15;
    const int khalf = lane >> 4;

    const int bm = blockIdx.y * BM;
    const int bn = blockIdx.x * BN;

    const unsigned ldsA0 = (unsigned)(uintptr_t)&sA[0][0];
    const unsigned ldsA1 = (unsigned)(uintptr_t)&sA[1][0];
    const unsigned ldsB0 = (unsigned)(uintptr_t)&sB[0][0];
    const unsigned ldsB1 = (unsigned)(uintptr_t)&sB[1][0];

    v8f acc[4][2] = {};

    const int nk = K / BK;
    if (wid == 0) {   // scalar-branch: only wave 0 issues the DMA descriptors
        tdm_load_2d(ldsA0, A + (size_t)bm * lda, K, M, lda, BK, BM);
        tdm_load_2d(ldsB0, Bt + (size_t)bn * ldb, K, N, ldb, BK, BN);
    }

    for (int t = 0; t < nk; ++t) {
        const int buf = t & 1;
        if (wid == 0) {
            if (t + 1 < nk) {
                const int kk = (t + 1) * BK;
                tdm_load_2d(buf ? ldsA0 : ldsA1, A + (size_t)bm * lda + kk, K, M, lda, BK, BM);
                tdm_load_2d(buf ? ldsB0 : ldsB1, Bt + (size_t)bn * ldb + kk, K, N, ldb, BK, BN);
                __builtin_amdgcn_s_wait_tensorcnt(2);   // current tile pair done
            } else {
                __builtin_amdgcn_s_wait_tensorcnt(0);
            }
        }
        __syncthreads();                                // tile visible to all waves

        const unsigned short* pA = sA[buf];
        const unsigned short* pB = sB[buf];

        FragBF af[4], bfg[2];
        #pragma unroll
        for (int i = 0; i < 4; ++i) {
            int m = wm + i * 16 + l16;
            af[i].q[0] = *(const uint4*)&pA[m * LSTRIDE + khalf * 16];
            af[i].q[1] = *(const uint4*)&pA[m * LSTRIDE + khalf * 16 + 8];
        }
        #pragma unroll
        for (int j = 0; j < 2; ++j) {
            int n = wn + j * 16 + l16;
            const uint4* pb = (const uint4*)&pB[n * LSTRIDE + khalf * 16];
            bfg[j].q[0] = pb[0];
            bfg[j].q[1] = pb[1];
        }
        #pragma unroll
        for (int i = 0; i < 4; ++i)
            #pragma unroll
            for (int j = 0; j < 2; ++j)
                acc[i][j] = wmma_bf16(af[i].v, bfg[j].v, acc[i][j]);
        __syncthreads();                                // done reading this buf
    }

    #pragma unroll
    for (int i = 0; i < 4; ++i)
        #pragma unroll
        for (int j = 0; j < 2; ++j) {
            int   gn = bn + wn + j * 16 + l16;
            float bv = bias[gn];
            if (f32out) {
                float* C = (float*)Cout;
                #pragma unroll
                for (int r = 0; r < 8; ++r) {
                    int gm = bm + wm + i * 16 + r + 8 * khalf;
                    C[(size_t)gm * ldc + gn] = acc[i][j][r] + bv;
                }
            } else {
                unsigned short* C = (unsigned short*)Cout;
                #pragma unroll
                for (int r = 0; r < 8; ++r) {
                    int gm = bm + wm + i * 16 + r + 8 * khalf;
                    C[(size_t)gm * ldc + gn] = f2bf(acc[i][j][r] + bv);
                }
            }
        }
}

// ===========================================================================
// Pooled K/V (bf16 in, bf16 out):
//   Kg[b][h][w][d] = sum_n k[b*64+w][h][n][d]*pool_w[n] + pool_b
// ===========================================================================
__global__ __launch_bounds__(256)
void pool_kv(const unsigned short* __restrict__ qkv, const float* __restrict__ pool_w,
             const float* __restrict__ pool_b,
             unsigned short* __restrict__ Kg, unsigned short* __restrict__ Vg)
{
    int t = blockIdx.x * blockDim.x + threadIdx.x;
    if (t >= 16 * 16 * 64 * 32) return;
    int d = t & 31;
    int w = (t >> 5) & 63;
    int h = (t >> 11) & 15;
    int b = t >> 15;
    size_t base = (size_t)(b * 64 + w) * 64 * 1536 + h * 32 + d;
    float aK = 0.f, aV = 0.f;
    #pragma unroll 8
    for (int n = 0; n < 64; ++n) {
        float pw = pool_w[n];
        aK += bf2f(qkv[base + (size_t)n * 1536 + 512])  * pw;
        aV += bf2f(qkv[base + (size_t)n * 1536 + 1024]) * pw;
    }
    float pb = pool_b[0];
    Kg[t] = f2bf(aK + pb);
    Vg[t] = f2bf(aV + pb);
}

// ===========================================================================
// Attention: one wave per (window b_, head h).  64 q x 64 keys, hd=32, bf16.
// mode 0: fine (K/V = window tokens); mode 1: coarse (pooled Kg/Vg).
// Bias index identical for both tables: (dy+7)*15 + (dx+7).
// ===========================================================================
__global__ __launch_bounds__(32)
void attn(const unsigned short* __restrict__ qkv, const unsigned short* __restrict__ Kg,
          const unsigned short* __restrict__ Vg, const float* __restrict__ bias_table,
          unsigned short* __restrict__ out, int mode)
{
    __shared__ unsigned short sP [64 * 72];  // probs bf16, row = query
    __shared__ unsigned short sVt[32 * 72];  // V^T bf16: [d][token]

    const int lane  = threadIdx.x;
    const int l16   = lane & 15;
    const int khalf = lane >> 4;
    const int h     = blockIdx.x & 15;
    const int b_    = blockIdx.x >> 4;

    const unsigned short* qb = qkv + (size_t)b_ * 64 * 1536 + h * 32;
    const unsigned short* kb;
    const unsigned short* vb;
    int ldkv;
    if (mode == 0) { kb = qb + 512; vb = qb + 1024; ldkv = 1536; }
    else {
        int b = b_ >> 6;
        size_t off = (size_t)(b * 16 + h) * 64 * 32;
        kb = Kg + off; vb = Vg + off; ldkv = 32;
    }

    // K as B fragments (bf16 direct from global; elems 0..15 = K khalf*16..+15)
    FragBF kf[4];
    #pragma unroll
    for (int j = 0; j < 4; ++j) {
        int n = j * 16 + l16;
        const unsigned short* src = kb + (size_t)n * ldkv + khalf * 16;
        kf[j].q[0] = *(const uint4*)(src + 0);
        kf[j].q[1] = *(const uint4*)(src + 8);
    }

    // stage V transposed into LDS
    #pragma unroll
    for (int tt = 0; tt < 2; ++tt) {
        int t = lane + tt * 32;
        const unsigned short* src = vb + (size_t)t * ldkv;
        #pragma unroll
        for (int dv = 0; dv < 32; ++dv)
            sVt[dv * 72 + t] = src[dv];
    }

    const float scale = 0.17677669529663687f;  // hd^-0.5

    for (int i = 0; i < 4; ++i) {
        // q as A fragment: elems 0..15 = K khalf*16..+15
        FragBF qf;
        {
            int m = i * 16 + l16;
            const unsigned short* qs = qb + (size_t)m * 1536 + khalf * 16;
            qf.q[0] = *(const uint4*)(qs + 0);
            qf.q[1] = *(const uint4*)(qs + 8);
        }

        v8f s[4] = {};
        #pragma unroll
        for (int j = 0; j < 4; ++j)
            s[j] = wmma_bf16(qf.v, kf[j].v, s[j]);

        // scale + relative-position bias
        #pragma unroll
        for (int j = 0; j < 4; ++j) {
            int kt = j * 16 + l16;
            int ky = kt >> 3, kx = kt & 7;
            #pragma unroll
            for (int r = 0; r < 8; ++r) {
                int qt = i * 16 + r + 8 * khalf;
                int qy = qt >> 3, qx = qt & 7;
                int idx = (qy - ky + 7) * 15 + (qx - kx + 7);
                s[j][r] = s[j][r] * scale + bias_table[idx * 16 + h];
            }
        }

        // softmax over 64 keys (16-lane-group reductions, wave32)
        #pragma unroll
        for (int r = 0; r < 8; ++r) {
            float mx = fmaxf(fmaxf(s[0][r], s[1][r]), fmaxf(s[2][r], s[3][r]));
            #pragma unroll
            for (int msk = 8; msk; msk >>= 1)
                mx = fmaxf(mx, __shfl_xor(mx, msk, 32));
            float p[4], sum = 0.f;
            #pragma unroll
            for (int j = 0; j < 4; ++j) { p[j] = __expf(s[j][r] - mx); sum += p[j]; }
            #pragma unroll
            for (int msk = 8; msk; msk >>= 1)
                sum += __shfl_xor(sum, msk, 32);
            float inv = 1.0f / sum;
            int prow = i * 16 + r + 8 * khalf;
            #pragma unroll
            for (int j = 0; j < 4; ++j)
                sP[prow * 72 + j * 16 + l16] = f2bf(p[j] * inv);
        }
    }
    __syncthreads();

    // out = P(64x64) @ V(64x32)
    for (int i = 0; i < 4; ++i) {
        v8f o[2] = {};
        #pragma unroll
        for (int kc = 0; kc < 2; ++kc) {
            FragBF pf;
            int m = i * 16 + l16;
            int kb0 = kc * 32 + khalf * 16;
            pf.q[0] = *(const uint4*)&sP[m * 72 + kb0];
            pf.q[1] = *(const uint4*)&sP[m * 72 + kb0 + 8];
            #pragma unroll
            for (int dt = 0; dt < 2; ++dt) {
                FragBF vf;
                int n = dt * 16 + l16;
                const uint4* pv = (const uint4*)&sVt[n * 72 + kc * 32 + khalf * 16];
                vf.q[0] = pv[0];
                vf.q[1] = pv[1];
                o[dt] = wmma_bf16(pf.v, vf.v, o[dt]);
            }
        }
        #pragma unroll
        for (int dt = 0; dt < 2; ++dt)
            #pragma unroll
            for (int r = 0; r < 8; ++r) {
                int gm = b_ * 64 + i * 16 + r + 8 * khalf;
                int gn = h * 32 + dt * 16 + l16;
                out[(size_t)gm * 512 + gn] = f2bf(o[dt][r]);
            }
    }
}

// ===========================================================================
extern "C" void kernel_launch(void* const* d_in, const int* in_sizes, int n_in,
                              void* d_out, int out_size, void* d_ws, size_t ws_size,
                              hipStream_t stream) {
    const float* x              = (const float*)d_in[0];   // (65536,512)
    const float* qkv_w          = (const float*)d_in[1];   // (512,1536)
    const float* qkv_b          = (const float*)d_in[2];
    const float* local_bias_tbl = (const float*)d_in[3];
    const float* global_bias_tbl= (const float*)d_in[4];
    const float* local_proj_w   = (const float*)d_in[5];   // (512,512)
    const float* local_proj_b   = (const float*)d_in[6];
    const float* pool_w         = (const float*)d_in[7];
    const float* pool_b         = (const float*)d_in[8];
    const float* global_proj_w  = (const float*)d_in[9];   // (512,512)
    const float* global_proj_b  = (const float*)d_in[10];
    const float* pfc_w          = (const float*)d_in[11];  // (1024,512)
    const float* pfc_b          = (const float*)d_in[12];
    float* outp = (float*)d_out;                           // (65536,512) f32

    typedef unsigned short bfp;
    const size_t M = 65536;
    char* ws = (char*)d_ws;
    size_t off = 0;
    bfp* xbf    = (bfp*)(ws + off); off += M * 512 * 2;              //  64 MB
    bfp* qkvwt  = (bfp*)(ws + off); off += (size_t)1536 * 512 * 2;   // 1.5 MB
    bfp* lpwt   = (bfp*)(ws + off); off += (size_t)512 * 512 * 2;
    bfp* gpwt   = (bfp*)(ws + off); off += (size_t)512 * 512 * 2;
    bfp* pfcwt  = (bfp*)(ws + off); off += (size_t)512 * 1024 * 2;
    bfp* qkv    = (bfp*)(ws + off); off += M * 1536 * 2;             // 192 MB
    bfp* Kg     = (bfp*)(ws + off); off += (size_t)524288 * 2;
    bfp* Vg     = (bfp*)(ws + off); off += (size_t)524288 * 2;
    bfp* Fo     = (bfp*)(ws + off); off += M * 512 * 2;              //  64 MB
    bfp* Co     = (bfp*)(ws + off); off += M * 512 * 2;              //  64 MB
    bfp* concat = (bfp*)(ws + off); off += M * 1024 * 2;             // 128 MB

    // 0) one-time conversions: x -> bf16; weights -> bf16 transposed [N x K]
    cvt_bf16<<<(int)((M * 512 / 4 + 255) / 256), 256, 0, stream>>>(x, xbf, M * 512 / 4);
    transpose_cvt<<<(1536 * 512 + 255) / 256, 256, 0, stream>>>(qkv_w, qkvwt, 512, 1536);
    transpose_cvt<<<(512 * 512 + 255) / 256, 256, 0, stream>>>(local_proj_w, lpwt, 512, 512);
    transpose_cvt<<<(512 * 512 + 255) / 256, 256, 0, stream>>>(global_proj_w, gpwt, 512, 512);
    transpose_cvt<<<(1024 * 512 + 255) / 256, 256, 0, stream>>>(pfc_w, pfcwt, 1024, 512);

    // 1) QKV projection (bf16 out)
    gemm_bf16<<<dim3(1536 / BN, M / BM), 256, 0, stream>>>(
        xbf, qkvwt, qkv_b, qkv, (int)M, 1536, 512, 512, 512, 1536, 0);

    // 2) pooled global K/V
    pool_kv<<<(524288 + 255) / 256, 256, 0, stream>>>(qkv, pool_w, pool_b, Kg, Vg);

    // 3) fine + coarse attention
    attn<<<16384, 32, 0, stream>>>(qkv, Kg, Vg, local_bias_tbl, Fo, 0);
    attn<<<16384, 32, 0, stream>>>(qkv, Kg, Vg, global_bias_tbl, Co, 1);

    // 4) projections into concat halves (ldc=1024, bf16 out)
    gemm_bf16<<<dim3(512 / BN, M / BM), 256, 0, stream>>>(
        Fo, lpwt, local_proj_b, concat, (int)M, 512, 512, 512, 512, 1024, 0);
    gemm_bf16<<<dim3(512 / BN, M / BM), 256, 0, stream>>>(
        Co, gpwt, global_proj_b, concat + 512, (int)M, 512, 512, 512, 512, 1024, 0);

    // 5) fused projection -> f32 output
    gemm_bf16<<<dim3(512 / BN, M / BM), 256, 0, stream>>>(
        concat, pfcwt, pfc_b, outp, (int)M, 512, 1024, 1024, 1024, 512, 1);
}
